// TinyViTFusion_77369540870833
// MI455X (gfx1250) — matrix operand order repaired and checked by reading, not compiled, not run on hardware
//
#include <hip/hip_runtime.h>
#include <math.h>

// ---------------------------------------------------------------------------
// Types for CDNA5 WMMA (wave32, v_wmma_f32_16x16x32_bf16)
// ---------------------------------------------------------------------------
typedef __bf16 bf16_t;
typedef __attribute__((ext_vector_type(16))) __bf16 v16bf;
typedef __attribute__((ext_vector_type(8)))  float  v8f;

union BFrag {
    v16bf  v;
    bf16_t e[16];
    float4 q[2];
};

__device__ __forceinline__ v8f wmma_bf16(v16bf a, v16bf b, v8f c) {
    // (neg_a, A, neg_b, B, c_mod, C, reuse_a, reuse_b)
    return __builtin_amdgcn_wmma_f32_16x16x32_bf16(false, a, false, b, (short)0, c, false, false);
}

// A fragment: 16x32 bf16, row-major source with leading dim `lda`.
// Layout used here (not ISA table order): lanes 0-15 row M=lane, K runs [kh..kh+15] contiguous, kh = (lane>>4)*16.
// zeroHi: zero elements 8..15 of lanes>=16 (K 24..31) for hd=24 padding.
__device__ __forceinline__ v16bf a_frag(const bf16_t* base, int lda, int m0, int k0,
                                        int lane, bool zeroHi) {
    int row = m0 + (lane & 15);
    int kh  = (lane >> 4) << 4;
    const bf16_t* p = base + (size_t)row * lda + k0 + kh;
    BFrag f;
    f.q[0] = *(const float4*)(p);
    f.q[1] = *(const float4*)(p + 8);
    if (zeroHi && (lane & 16)) {
#pragma unroll
        for (int e = 8; e < 16; ++e) f.e[e] = (bf16_t)0.0f;
    }
    return f.v;
}

// B fragment where B = X^T and X is row-major [n][k] with leading dim `ld`
// (attention scores: B[k][n] = K[n][k]); contiguous 16 bf16 per lane.
__device__ __forceinline__ v16bf bt_frag(const bf16_t* base, int ld, int n0, int k0,
                                         int lane, bool zeroHi) {
    int nrow = n0 + (lane & 15);
    const bf16_t* p = base + (size_t)nrow * ld + k0 + ((lane >> 4) << 4);
    BFrag f;
    f.q[0] = *(const float4*)(p);
    f.q[1] = *(const float4*)(p + 8);
    if (zeroHi && (lane & 16)) {
#pragma unroll
        for (int e = 8; e < 16; ++e) f.e[e] = (bf16_t)0.0f;
    }
    return f.v;
}

// B fragment from row-major B[k][n] with leading dim `ld` (strided gather).
__device__ __forceinline__ v16bf b_strided(const bf16_t* base, int ld, int k0, int n0, int lane) {
    int col = n0 + (lane & 15);
    int kb  = k0 + ((lane >> 4) << 4);
    BFrag f;
#pragma unroll
    for (int e = 0; e < 16; ++e) f.e[e] = base[(size_t)(kb + e) * ld + col];
    return f.v;
}

// B fragment from pre-swizzled (packed) weights: contiguous 32B per lane.
__device__ __forceinline__ v16bf b_packed(const bf16_t* w, int kTiles, int nt, int kt, int lane) {
    const bf16_t* p = w + ((size_t)((nt * kTiles + kt) * 32 + lane) << 4);
    BFrag f;
    f.q[0] = *(const float4*)(p);
    f.q[1] = *(const float4*)(p + 8);
    return f.v;
}

// ---------------------------------------------------------------------------
// Weight packer: src is fp32 [N][K] row-major; dest is WMMA-B-fragment order.
// ---------------------------------------------------------------------------
__global__ __launch_bounds__(256) void k_pack(const float* __restrict__ w,
                                              bf16_t* __restrict__ o, int K, int N) {
    int idx = blockIdx.x * 256 + threadIdx.x;
    if (idx >= K * N) return;
    int e    = idx & 15;
    int lane = (idx >> 4) & 31;
    int t    = idx >> 9;
    int kT   = K >> 5;
    int kt   = t % kT;
    int nt   = t / kT;
    int col  = nt * 16 + (lane & 15);
    int kk   = kt * 32 + ((lane >> 4) << 4) + e;
    o[idx] = (bf16_t)w[(size_t)col * K + kk];
}

// ---------------------------------------------------------------------------
// Patch embed: stride-3 3x3 conv == exact im2col GEMM (M=B*128*128, K=288, N=96)
// ---------------------------------------------------------------------------
__global__ __launch_bounds__(256) void k_patch(const float* __restrict__ x,
                                               const bf16_t* __restrict__ Wp,
                                               const float* __restrict__ bp,
                                               float* __restrict__ feat) {
    __shared__ __align__(32) bf16_t sA[16 * 288];
    int tid = threadIdx.x, lane = tid & 31, wave = tid >> 5;
    int m0 = blockIdx.x * 16;
    for (int idx = tid; idx < 16 * 288; idx += 256) {
        int t = idx / 288, k = idx - t * 288;
        int m = m0 + t;
        int b = m >> 14, pix = m & 16383, oy = pix >> 7, ox = pix & 127;
        int c = k / 9, rem = k - c * 9, rr = rem / 3, ss = rem - rr * 3;
        sA[idx] = (bf16_t)x[((size_t)(b * 32 + c) * 384 + oy * 3 + rr) * 384 + ox * 3 + ss];
    }
    __syncthreads();
    for (int nt = wave; nt < 6; nt += 8) {
        v8f acc = {0.f, 0.f, 0.f, 0.f, 0.f, 0.f, 0.f, 0.f};
#pragma unroll
        for (int kt = 0; kt < 9; ++kt) {
            __builtin_prefetch(Wp + ((size_t)((nt * 9 + kt + 1) * 32) << 4), 0, 3);
            acc = wmma_bf16(a_frag(sA, 288, 0, kt * 32, lane, false),
                            b_packed(Wp, 9, nt, kt, lane), acc);
        }
        int col = nt * 16 + (lane & 15);
        float bias = bp[col];
        int kh = (lane >> 4) << 3;
#pragma unroll
        for (int i = 0; i < 8; ++i) {
            int m = m0 + i + kh;
            int b = m >> 14, pix = m & 16383, oy = pix >> 7, ox = pix & 127;
            feat[((size_t)(b * 96 + col) << 14) + (oy << 7) + ox] = acc[i] + bias;
        }
    }
}

// ---------------------------------------------------------------------------
// Depthwise 3x3 pad-1 positional conv + residual: out = in + dw(in) + bias
// ---------------------------------------------------------------------------
__global__ __launch_bounds__(256) void k_dw(const float* __restrict__ fin,
                                            const float* __restrict__ pw,
                                            const float* __restrict__ pb,
                                            float* __restrict__ fout) {
    int idx = blockIdx.x * 256 + threadIdx.x;  // B*96*128*128 = 6291456
    int xq = idx & 127, yq = (idx >> 7) & 127;
    int c = (idx >> 14) % 96;
    const float* plane = fin + (size_t)(idx & ~16383);
    float a = pb[c];
#pragma unroll
    for (int dy = -1; dy <= 1; ++dy)
#pragma unroll
        for (int dx = -1; dx <= 1; ++dx) {
            int yy = yq + dy, xx = xq + dx;
            if ((unsigned)yy < 128u && (unsigned)xx < 128u)
                a += pw[c * 9 + (dy + 1) * 3 + (dx + 1)] * plane[(yy << 7) + xx];
        }
    fout[idx] = fin[idx] + a;
}

// ---------------------------------------------------------------------------
// Fused Swin block: one workgroup per 8x8 window (64 tokens x 96 ch).
// Residual stream lives in registers (24 f32/thread); LDS is a 64KB arena.
// ---------------------------------------------------------------------------
__global__ __launch_bounds__(256) void k_swin(
    const float* __restrict__ featIn, float* __restrict__ featOut,
    const float* __restrict__ ln1g, const float* __restrict__ ln1b,
    const bf16_t* __restrict__ qkvW, const float* __restrict__ qkvB,
    const bf16_t* __restrict__ projW, const float* __restrict__ projB,
    const float* __restrict__ ln2g, const float* __restrict__ ln2b,
    const bf16_t* __restrict__ fc1W, const float* __restrict__ fc1B,
    const bf16_t* __restrict__ fc2W, const float* __restrict__ fc2B,
    int shift) {
    __shared__ __align__(32) char smem[65536];
    bf16_t* s_ln  = (bf16_t*)(smem);           // [64][96]  bf16  12288B
    bf16_t* s_qkv = (bf16_t*)(smem + 12288);   // [64][288] bf16  36864B
    bf16_t* s_sb  = (bf16_t*)(smem + 49152);   // [64][64]  bf16   8192B (scores)
    bf16_t* s_p   = (bf16_t*)(smem + 57344);   // [64][64]  bf16   8192B (softmax P)
    float*  s_f32 = (float*)(smem + 12288);    // [64][96]  f32   (proj out, overlays qkv)
    bf16_t* s_h   = (bf16_t*)(smem + 12288);   // [64][192] bf16  (fc1 out, overlays qkv)
    float*  s_f2  = (float*)(smem + 36864);    // [64][96]  f32   (fc2 out, overlays tail)

    int tid = threadIdx.x, lane = tid & 31, wave = tid >> 5;
    int blk = blockIdx.x;
    int b = blk >> 8, w = blk & 255, wy = w >> 4, wx = w & 15;
    int tok = tid >> 2, part = tid & 3, ch0 = part * 24;
    int ty = tok >> 3, tx = tok & 7;
    int py = (wy * 8 + ty + shift) & 127;
    int px = (wx * 8 + tx + shift) & 127;
    size_t pixOff = ((size_t)b * 96 << 14) + (py << 7) + px;

    float r[24];
#pragma unroll
    for (int j = 0; j < 24; ++j) r[j] = featIn[pixOff + ((size_t)(ch0 + j) << 14)];

    auto layernorm = [&](const float* g, const float* bb) {
        float s = 0.f, s2 = 0.f;
#pragma unroll
        for (int j = 0; j < 24; ++j) { s += r[j]; s2 += r[j] * r[j]; }
        s  += __shfl_xor(s, 1);  s  += __shfl_xor(s, 2);
        s2 += __shfl_xor(s2, 1); s2 += __shfl_xor(s2, 2);
        float m = s * (1.f / 96.f);
        float v = s2 * (1.f / 96.f) - m * m;
        float rs = rsqrtf(v + 1e-5f);
#pragma unroll
        for (int j = 0; j < 24; ++j)
            s_ln[tok * 96 + ch0 + j] = (bf16_t)((r[j] - m) * rs * g[ch0 + j] + bb[ch0 + j]);
    };

    // ---- LN1 ----
    layernorm(ln1g, ln1b);
    __syncthreads();

    // ---- QKV GEMM: (64x96)@(96x288), K=3 steps ----
    for (int t = wave; t < 72; t += 8) {
        int mt = t / 18, nt = t % 18;
        v8f acc = {0.f, 0.f, 0.f, 0.f, 0.f, 0.f, 0.f, 0.f};
#pragma unroll
        for (int kt = 0; kt < 3; ++kt) {
            acc = wmma_bf16(a_frag(s_ln, 96, mt * 16, kt * 32, lane, false),
                            b_packed(qkvW, 3, nt, kt, lane), acc);
        }
        int col = nt * 16 + (lane & 15);
        float bias = qkvB[col];
        float sc = (col < 96) ? 0.20412414523193154f : 1.f;  // hd^-0.5, hd=24
        int kh = (lane >> 4) << 3;
#pragma unroll
        for (int i = 0; i < 8; ++i)
            s_qkv[(mt * 16 + i + kh) * 288 + col] = (bf16_t)((acc[i] + bias) * sc);
    }
    __syncthreads();

    // ---- attention: 4 heads, hd=24 zero-padded to K=32 ----
    for (int h = 0; h < 4; ++h) {
        // S = Q K^T : 4x4 tiles, single k-step
        for (int t = wave; t < 16; t += 8) {
            int mt = t >> 2, nt = t & 3;
            v8f acc = {0.f, 0.f, 0.f, 0.f, 0.f, 0.f, 0.f, 0.f};
            acc = wmma_bf16(a_frag(s_qkv + h * 24, 288, mt * 16, 0, lane, true),
                            bt_frag(s_qkv + 96 + h * 24, 288, nt * 16, 0, lane, true), acc);
            int col = nt * 16 + (lane & 15);
            int kh = (lane >> 4) << 3;
#pragma unroll
            for (int i = 0; i < 8; ++i)
                s_sb[(mt * 16 + i + kh) * 64 + col] = (bf16_t)acc[i];
        }
        __syncthreads();
        // row softmax (one thread per row)
        if (tid < 64) {
            int row = tid;
            float mx = -1e30f;
            for (int j = 0; j < 64; ++j) mx = fmaxf(mx, (float)s_sb[row * 64 + j]);
            float sum = 0.f;
            for (int j = 0; j < 64; ++j) sum += __expf((float)s_sb[row * 64 + j] - mx);
            float inv = 1.f / sum;
            for (int j = 0; j < 64; ++j)
                s_p[row * 64 + j] = (bf16_t)(__expf((float)s_sb[row * 64 + j] - mx) * inv);
        }
        __syncthreads();
        // O = P V : 4x2 tiles, K=64 (2 steps); write into s_ln[:, h*24..]
        for (int t = wave; t < 8; t += 8) {
            int mt = t >> 1, nt = t & 1;
            v8f acc = {0.f, 0.f, 0.f, 0.f, 0.f, 0.f, 0.f, 0.f};
#pragma unroll
            for (int kt = 0; kt < 2; ++kt)
                acc = wmma_bf16(a_frag(s_p, 64, mt * 16, kt * 32, lane, false),
                                b_strided(s_qkv + 192 + h * 24, 288, kt * 32, nt * 16, lane), acc);
            int c = nt * 16 + (lane & 15);
            int kh = (lane >> 4) << 3;
            if (c < 24) {
#pragma unroll
                for (int i = 0; i < 8; ++i)
                    s_ln[(mt * 16 + i + kh) * 96 + h * 24 + c] = (bf16_t)acc[i];
            }
        }
        __syncthreads();
    }

    // ---- proj: (64x96)@(96x96) -> f32 scratch ----
    for (int t = wave; t < 24; t += 8) {
        int mt = t / 6, nt = t % 6;
        v8f acc = {0.f, 0.f, 0.f, 0.f, 0.f, 0.f, 0.f, 0.f};
#pragma unroll
        for (int kt = 0; kt < 3; ++kt)
            acc = wmma_bf16(a_frag(s_ln, 96, mt * 16, kt * 32, lane, false),
                            b_packed(projW, 3, nt, kt, lane), acc);
        int col = nt * 16 + (lane & 15);
        int kh = (lane >> 4) << 3;
#pragma unroll
        for (int i = 0; i < 8; ++i) s_f32[(mt * 16 + i + kh) * 96 + col] = acc[i];
    }
    __syncthreads();
#pragma unroll
    for (int j = 0; j < 24; ++j) r[j] += s_f32[tok * 96 + ch0 + j] + projB[ch0 + j];

    // ---- LN2 ----
    layernorm(ln2g, ln2b);
    __syncthreads();

    // ---- fc1 + SiLU: (64x96)@(96x192) ----
    for (int t = wave; t < 48; t += 8) {
        int mt = t / 12, nt = t % 12;
        v8f acc = {0.f, 0.f, 0.f, 0.f, 0.f, 0.f, 0.f, 0.f};
#pragma unroll
        for (int kt = 0; kt < 3; ++kt)
            acc = wmma_bf16(a_frag(s_ln, 96, mt * 16, kt * 32, lane, false),
                            b_packed(fc1W, 3, nt, kt, lane), acc);
        int n = nt * 16 + (lane & 15);
        float bias = fc1B[n];
        int kh = (lane >> 4) << 3;
#pragma unroll
        for (int i = 0; i < 8; ++i) {
            float v = acc[i] + bias;
            v = v / (1.f + __expf(-v));  // silu
            s_h[(mt * 16 + i + kh) * 192 + n] = (bf16_t)v;
        }
    }
    __syncthreads();

    // ---- fc2: (64x192)@(192x96) ----
    for (int t = wave; t < 24; t += 8) {
        int mt = t / 6, nt = t % 6;
        v8f acc = {0.f, 0.f, 0.f, 0.f, 0.f, 0.f, 0.f, 0.f};
#pragma unroll
        for (int kt = 0; kt < 6; ++kt)
            acc = wmma_bf16(a_frag(s_h, 192, mt * 16, kt * 32, lane, false),
                            b_packed(fc2W, 6, nt, kt, lane), acc);
        int col = nt * 16 + (lane & 15);
        int kh = (lane >> 4) << 3;
#pragma unroll
        for (int i = 0; i < 8; ++i) s_f2[(mt * 16 + i + kh) * 96 + col] = acc[i];
    }
    __syncthreads();
#pragma unroll
    for (int j = 0; j < 24; ++j) r[j] += s_f2[tok * 96 + ch0 + j] + fc2B[ch0 + j];

#pragma unroll
    for (int j = 0; j < 24; ++j) featOut[pixOff + ((size_t)(ch0 + j) << 14)] = r[j];
}

// ---------------------------------------------------------------------------
// Head: bilinear(128->384) resize computed on the fly, conv1(3x3,96->96) as
// im2col WMMA GEMM (K=864). mode 0: per-sample sum/sumsq stats.
// mode 1: recompute, normalize (per-sample), affine, SiLU, conv2 (1x1, WMMA).
// ---------------------------------------------------------------------------
__device__ __forceinline__ float bilin128(const float* f, int b, int c, int ry, int rx) {
    float fy = (ry + 0.5f) * (1.0f / 3.0f) - 0.5f;
    float fx = (rx + 0.5f) * (1.0f / 3.0f) - 0.5f;
    int y0 = (int)floorf(fy); float tyf = fy - y0;
    int x0 = (int)floorf(fx); float txf = fx - x0;
    int y1 = min(max(y0 + 1, 0), 127); y0 = min(max(y0, 0), 127);
    int x1 = min(max(x0 + 1, 0), 127); x0 = min(max(x0, 0), 127);
    const float* p = f + ((size_t)(b * 96 + c) << 14);
    float v00 = p[(y0 << 7) + x0], v01 = p[(y0 << 7) + x1];
    float v10 = p[(y1 << 7) + x0], v11 = p[(y1 << 7) + x1];
    float a = v00 + tyf * (v10 - v00);
    float bb = v01 + tyf * (v11 - v01);
    return a + txf * (bb - a);
}

__global__ __launch_bounds__(256) void k_head(const float* __restrict__ feat,
                                              const bf16_t* __restrict__ W1,
                                              const float* __restrict__ b1,
                                              const float* __restrict__ gng,
                                              const float* __restrict__ gnb,
                                              const bf16_t* __restrict__ W2,
                                              const float* __restrict__ b2,
                                              float* __restrict__ stats,
                                              float* __restrict__ out, int mode) {
    __shared__ __align__(32) bf16_t sA[16 * 864];
    __shared__ __align__(32) bf16_t sAct[16 * 96];
    __shared__ float sRed[2];
    int tid = threadIdx.x, lane = tid & 31, wave = tid >> 5;
    int m0 = blockIdx.x * 16;
    int b = m0 / 147456;
    if (tid < 2) sRed[tid] = 0.f;

    for (int idx = tid; idx < 16 * 864; idx += 256) {
        int t = idx / 864, k = idx - t * 864;
        int m = m0 + t;
        int pix = m - b * 147456, oy = pix / 384, ox = pix - oy * 384;
        int c = k / 9, rem = k - c * 9, rr = rem / 3, ss = rem - rr * 3;
        int ry = oy - 1 + rr, rx = ox - 1 + ss;
        float v = 0.f;
        if ((unsigned)ry < 384u && (unsigned)rx < 384u) v = bilin128(feat, b, c, ry, rx);
        sA[idx] = (bf16_t)v;
    }
    __syncthreads();

    float mu = 0.f, inv = 1.f;
    if (mode) {
        float s1 = stats[b * 2], s2 = stats[b * 2 + 1];
        const float cnt = 96.f * 147456.f;
        mu = s1 / cnt;
        float var = s2 / cnt - mu * mu;
        inv = rsqrtf(var + 1e-5f);
    }

    for (int nt = wave; nt < 6; nt += 8) {
        v8f acc = {0.f, 0.f, 0.f, 0.f, 0.f, 0.f, 0.f, 0.f};
#pragma unroll
        for (int kt = 0; kt < 27; ++kt) {
            __builtin_prefetch(W1 + ((size_t)((nt * 27 + kt + 1) * 32) << 4), 0, 3);
            acc = wmma_bf16(a_frag(sA, 864, 0, kt * 32, lane, false),
                            b_packed(W1, 27, nt, kt, lane), acc);
        }
        int col = nt * 16 + (lane & 15);
        float bias = b1[col];
        int kh = (lane >> 4) << 3;
        if (mode == 0) {
            float ls = 0.f, ls2 = 0.f;
#pragma unroll
            for (int i = 0; i < 8; ++i) {
                float v = acc[i] + bias;
                ls += v; ls2 += v * v;
            }
            atomicAdd(&sRed[0], ls);
            atomicAdd(&sRed[1], ls2);
        } else {
            float g = gng[col], gb = gnb[col];
#pragma unroll
            for (int i = 0; i < 8; ++i) {
                float v = acc[i] + bias;
                v = (v - mu) * inv * g + gb;
                v = v / (1.f + __expf(-v));  // silu
                sAct[(i + kh) * 96 + col] = (bf16_t)v;
            }
        }
    }
    __syncthreads();

    if (mode == 0) {
        if (tid == 0) {
            atomicAdd(&stats[b * 2], sRed[0]);
            atomicAdd(&stats[b * 2 + 1], sRed[1]);
        }
    } else {
        // conv2: 1x1, (16x96)@(96x32) WMMA, straight to output
        for (int nt = wave; nt < 2; nt += 8) {
            v8f acc = {0.f, 0.f, 0.f, 0.f, 0.f, 0.f, 0.f, 0.f};
#pragma unroll
            for (int kt = 0; kt < 3; ++kt)
                acc = wmma_bf16(a_frag(sAct, 96, 0, kt * 32, lane, false),
                                b_packed(W2, 3, nt, kt, lane), acc);
            int oc = nt * 16 + (lane & 15);
            float bias = b2[oc];
            int kh = (lane >> 4) << 3;
#pragma unroll
            for (int i = 0; i < 8; ++i) {
                int m = m0 + i + kh;
                int pix = m - b * 147456;
                out[(size_t)(b * 32 + oc) * 147456 + pix] = acc[i] + bias;
            }
        }
    }
}

// ---------------------------------------------------------------------------
// Host orchestration
// ---------------------------------------------------------------------------
extern "C" void kernel_launch(void* const* d_in, const int* in_sizes, int n_in,
                              void* d_out, int out_size, void* d_ws, size_t ws_size,
                              hipStream_t stream) {
    (void)in_sizes; (void)n_in; (void)out_size; (void)ws_size;
    const float* x        = (const float*)d_in[0];
    const float* patch_w  = (const float*)d_in[1];
    const float* patch_b  = (const float*)d_in[2];
    const float* pos_w    = (const float*)d_in[3];
    const float* pos_b    = (const float*)d_in[4];
    const float* ln1_g    = (const float*)d_in[5];
    const float* ln1_b    = (const float*)d_in[6];
    const float* qkv_w    = (const float*)d_in[7];
    const float* qkv_b    = (const float*)d_in[8];
    const float* proj_w   = (const float*)d_in[9];
    const float* proj_b   = (const float*)d_in[10];
    const float* ln2_g    = (const float*)d_in[11];
    const float* ln2_b    = (const float*)d_in[12];
    const float* fc1_w    = (const float*)d_in[13];
    const float* fc1_b    = (const float*)d_in[14];
    const float* fc2_w    = (const float*)d_in[15];
    const float* fc2_b    = (const float*)d_in[16];
    const float* conv1_w  = (const float*)d_in[17];
    const float* conv1_b  = (const float*)d_in[18];
    const float* gn_g     = (const float*)d_in[19];
    const float* gn_b     = (const float*)d_in[20];
    const float* conv2_w  = (const float*)d_in[21];
    const float* conv2_b  = (const float*)d_in[22];
    float* outp = (float*)d_out;

    char* p = (char*)d_ws;
    float* featA = (float*)p; p += (size_t)4 * 96 * 128 * 128 * 4;
    float* featB = (float*)p; p += (size_t)4 * 96 * 128 * 128 * 4;
    float* stats = (float*)p; p += 256;
    bf16_t* pkPatch = (bf16_t*)p; p += (size_t)288 * 96 * 2;
    bf16_t* pkQKV0  = (bf16_t*)p; p += (size_t)96 * 288 * 2;
    bf16_t* pkQKV1  = (bf16_t*)p; p += (size_t)96 * 288 * 2;
    bf16_t* pkProj0 = (bf16_t*)p; p += (size_t)96 * 96 * 2;
    bf16_t* pkProj1 = (bf16_t*)p; p += (size_t)96 * 96 * 2;
    bf16_t* pkFc10  = (bf16_t*)p; p += (size_t)96 * 192 * 2;
    bf16_t* pkFc11  = (bf16_t*)p; p += (size_t)96 * 192 * 2;
    bf16_t* pkFc20  = (bf16_t*)p; p += (size_t)192 * 96 * 2;
    bf16_t* pkFc21  = (bf16_t*)p; p += (size_t)192 * 96 * 2;
    bf16_t* pkC1    = (bf16_t*)p; p += (size_t)864 * 96 * 2;
    bf16_t* pkC2    = (bf16_t*)p; p += (size_t)96 * 32 * 2;

    hipMemsetAsync(stats, 0, 8 * sizeof(float), stream);

    // weight packing (deterministic every call)
    k_pack<<<(288 * 96) / 256, 256, 0, stream>>>(patch_w, pkPatch, 288, 96);
    k_pack<<<(96 * 288) / 256, 256, 0, stream>>>(qkv_w, pkQKV0, 96, 288);
    k_pack<<<(96 * 288) / 256, 256, 0, stream>>>(qkv_w + 288 * 96, pkQKV1, 96, 288);
    k_pack<<<(96 * 96) / 256, 256, 0, stream>>>(proj_w, pkProj0, 96, 96);
    k_pack<<<(96 * 96) / 256, 256, 0, stream>>>(proj_w + 96 * 96, pkProj1, 96, 96);
    k_pack<<<(96 * 192) / 256, 256, 0, stream>>>(fc1_w, pkFc10, 96, 192);
    k_pack<<<(96 * 192) / 256, 256, 0, stream>>>(fc1_w + 192 * 96, pkFc11, 96, 192);
    k_pack<<<(192 * 96) / 256, 256, 0, stream>>>(fc2_w, pkFc20, 192, 96);
    k_pack<<<(192 * 96) / 256, 256, 0, stream>>>(fc2_w + 96 * 192, pkFc21, 192, 96);
    k_pack<<<(864 * 96) / 256, 256, 0, stream>>>(conv1_w, pkC1, 864, 96);
    k_pack<<<(96 * 32) / 256, 256, 0, stream>>>(conv2_w, pkC2, 96, 32);

    // patch embed + positional depthwise conv
    k_patch<<<4096, 256, 0, stream>>>(x, pkPatch, patch_b, featA);
    k_dw<<<24576, 256, 0, stream>>>(featA, pos_w, pos_b, featB);

    // two swin blocks (block 1 shifted by 4)
    k_swin<<<1024, 256, 0, stream>>>(featB, featA, ln1_g, ln1_b, pkQKV0, qkv_b,
                                     pkProj0, proj_b, ln2_g, ln2_b, pkFc10, fc1_b,
                                     pkFc20, fc2_b, 0);
    k_swin<<<1024, 256, 0, stream>>>(featA, featB, ln1_g + 96, ln1_b + 96, pkQKV1,
                                     qkv_b + 288, pkProj1, proj_b + 96, ln2_g + 96,
                                     ln2_b + 96, pkFc11, fc1_b + 192, pkFc21,
                                     fc2_b + 96, 4);

    // head: stats pass then normalize+silu+conv2 pass (conv1 recomputed)
    k_head<<<36864, 256, 0, stream>>>(featB, pkC1, conv1_b, gn_g, gn_b, pkC2, conv2_b,
                                      stats, outp, 0);
    k_head<<<36864, 256, 0, stream>>>(featB, pkC1, conv1_b, gn_g, gn_b, pkC2, conv2_b,
                                      stats, outp, 1);
}
